// SSM_28381143891999
// MI455X (gfx1250) — matrix operand order repaired and checked by reading, not compiled, not run on hardware
//
#include <hip/hip_runtime.h>
#include <hip/hip_bf16.h>

typedef __bf16 bf16_t;
typedef __attribute__((ext_vector_type(16))) __bf16 v16bf;
typedef __attribute__((ext_vector_type(8)))  __bf16 v8bf;
typedef __attribute__((ext_vector_type(4)))  __bf16 v4bf;
typedef __attribute__((ext_vector_type(8)))  float  v8f;
typedef __attribute__((ext_vector_type(4)))  int    v4i;

#define TM 128
#define TN 128
#define TK 32
#define LDT 40          // padded LDS row stride (bf16); 80 B = 16B multiple
#define CHUNK 128       // scan chunk length (divides T=2048; batch edges align)

#if defined(__gfx1250__) && __has_builtin(__builtin_amdgcn_global_load_async_to_lds_b128)
#define ASYNC_LDS 1
#else
#define ASYNC_LDS 0
#endif

__device__ __forceinline__ void wait_async0() {
#if ASYNC_LDS
#if __has_builtin(__builtin_amdgcn_s_wait_asynccnt)
  __builtin_amdgcn_s_wait_asynccnt(0);
#else
  asm volatile("s_wait_asynccnt 0x0" ::: "memory");
#endif
#endif
}

#if ASYNC_LDS
__device__ __forceinline__ void async_b128(const bf16_t* g, bf16_t* l) {
  __builtin_amdgcn_global_load_async_to_lds_b128(
      (__attribute__((address_space(1))) v4i*)(g),
      (__attribute__((address_space(3))) v4i*)(l), 0, 0);
}
#endif

__device__ __forceinline__ v16bf cat8(v8bf lo, v8bf hi) {
  return __builtin_shufflevector(lo, hi, 0, 1, 2, 3, 4, 5, 6, 7, 8, 9, 10, 11,
                                 12, 13, 14, 15);
}

// ---------------------------------------------------------------------------
// Prep kernels
// ---------------------------------------------------------------------------
__global__ void f32_to_bf16_vec(const float4* __restrict__ in,
                                v4bf* __restrict__ outp, long n4) {
  long gid = (long)blockIdx.x * blockDim.x + threadIdx.x;
  if (gid < n4) {
    float4 v = in[gid];
    outp[gid] = v4bf{(bf16_t)v.x, (bf16_t)v.y, (bf16_t)v.z, (bf16_t)v.w};
  }
}

// Wt[n][k] = bf16(W[k][n]); LDS-tiled so both sides are coalesced.
__global__ __launch_bounds__(256) void transpose_to_bf16(
    const float* __restrict__ W, bf16_t* __restrict__ Wt, int K, int N) {
  __shared__ bf16_t tile[64][65];
  int bk = blockIdx.x * 64, bn = blockIdx.y * 64;
#pragma unroll
  for (int i = 0; i < 16; i++) {
    int idx = threadIdx.x + i * 256;   // 4096 elements
    int r = idx >> 6, c = idx & 63;    // r: k-offset, c: n-offset (coalesced)
    tile[c][r] = (bf16_t)W[(size_t)(bk + r) * N + bn + c];
  }
  __syncthreads();
#pragma unroll
  for (int i = 0; i < 16; i++) {
    int idx = threadIdx.x + i * 256;
    int r = idx >> 6, c = idx & 63;    // r: n-offset, c: k-offset (coalesced)
    Wt[(size_t)(bn + r) * K + bk + c] = tile[r][c];
  }
}

__global__ void prep_A(const float* __restrict__ logA, float* __restrict__ Ach,
                       int d) {
  int gid = blockIdx.x * blockDim.x + threadIdx.x;
  if (gid < d) Ach[gid] = -__expf(logA[gid]);
}

// ---------------------------------------------------------------------------
// bf16 WMMA GEMM. 128x128 tile, 8 waves, TK=32, double-buffered LDS with one
// barrier per K-step. Next tile is staged via async-to-LDS (ASYNCcnt) when the
// toolchain exposes the builtin, else via register prefetch.
//   EPI 0: out bf16 = (acc + b_in[n]) * B_vec[n]
//   EPI 1: out f32  =  acc + x[m][n] * D[n]
// A: bf16 [M][K].  B: pre-transposed bf16 [N][K].
// ---------------------------------------------------------------------------
template <int EPI>
__global__ __launch_bounds__(256) void ssm_gemm(
    const bf16_t* __restrict__ Ag, const bf16_t* __restrict__ Bt,
    const float* __restrict__ ep0, const float* __restrict__ ep1,
    void* __restrict__ outv, int M, int N, int K) {
  __shared__ bf16_t sT[2 * (TM + TN) * LDT];   // 40960 B

  const int tid  = threadIdx.x;
  const int lane = tid & 31;
  const int wave = tid >> 5;
  const int wm   = wave & 3;   // 4 waves along M (32 rows each)
  const int wn   = wave >> 2;  // 2 waves along N (64 cols each)
  const int half = lane >> 4;
  const int l16  = lane & 15;

  const int mBase = blockIdx.y * TM;
  const int nBase = blockIdx.x * TN;

  // staging decomposition: each (A|B) tile = 128 rows x 32 bf16 = 512 x 16B
  const int sr0 = (tid + 0)   >> 2, sc0 = ((tid + 0)   & 3) << 3;
  const int sr1 = (tid + 256) >> 2, sc1 = ((tid + 256) & 3) << 3;

  v8f acc[2][4];
#pragma unroll
  for (int i = 0; i < 2; i++)
#pragma unroll
    for (int j = 0; j < 4; j++)
      acc[i][j] = v8f{0.f, 0.f, 0.f, 0.f, 0.f, 0.f, 0.f, 0.f};

  int cur = 0;

  // ---- prologue: stage tile 0 into buffer 0 ----
  {
    bf16_t* nb = sT;
#if ASYNC_LDS
    async_b128(Ag + (size_t)(mBase + sr0) * K + sc0, nb + sr0 * LDT + sc0);
    async_b128(Ag + (size_t)(mBase + sr1) * K + sc1, nb + sr1 * LDT + sc1);
    async_b128(Bt + (size_t)(nBase + sr0) * K + sc0, nb + (TM + sr0) * LDT + sc0);
    async_b128(Bt + (size_t)(nBase + sr1) * K + sc1, nb + (TM + sr1) * LDT + sc1);
#else
    v8bf a0 = *(const v8bf*)(Ag + (size_t)(mBase + sr0) * K + sc0);
    v8bf a1 = *(const v8bf*)(Ag + (size_t)(mBase + sr1) * K + sc1);
    v8bf b0 = *(const v8bf*)(Bt + (size_t)(nBase + sr0) * K + sc0);
    v8bf b1 = *(const v8bf*)(Bt + (size_t)(nBase + sr1) * K + sc1);
    *(v8bf*)(nb + sr0 * LDT + sc0) = a0;
    *(v8bf*)(nb + sr1 * LDT + sc1) = a1;
    *(v8bf*)(nb + (TM + sr0) * LDT + sc0) = b0;
    *(v8bf*)(nb + (TM + sr1) * LDT + sc1) = b1;
#endif
  }

  for (int k0 = 0; k0 < K; k0 += TK) {
    wait_async0();       // my async loads into buf[cur] have landed
    __syncthreads();     // everyone's tile in buf[cur] ready; buf[cur^1] free

    const bool more = (k0 + TK) < K;
    bf16_t* nb = sT + (cur ^ 1) * ((TM + TN) * LDT);
    const int kn = k0 + TK;

#if ASYNC_LDS
    if (more) {          // issue next tile; DMA runs under the WMMAs below
      async_b128(Ag + (size_t)(mBase + sr0) * K + kn + sc0, nb + sr0 * LDT + sc0);
      async_b128(Ag + (size_t)(mBase + sr1) * K + kn + sc1, nb + sr1 * LDT + sc1);
      async_b128(Bt + (size_t)(nBase + sr0) * K + kn + sc0, nb + (TM + sr0) * LDT + sc0);
      async_b128(Bt + (size_t)(nBase + sr1) * K + kn + sc1, nb + (TM + sr1) * LDT + sc1);
    }
#else
    v8bf a0, a1, b0, b1;
    if (more) {          // issue loads; waited only at the ds_store below
      a0 = *(const v8bf*)(Ag + (size_t)(mBase + sr0) * K + kn + sc0);
      a1 = *(const v8bf*)(Ag + (size_t)(mBase + sr1) * K + kn + sc1);
      b0 = *(const v8bf*)(Bt + (size_t)(nBase + sr0) * K + kn + sc0);
      b1 = *(const v8bf*)(Bt + (size_t)(nBase + sr1) * K + kn + sc1);
    }
#endif

    // ---- compute on buf[cur] ----
    const bf16_t* bufA = sT + cur * ((TM + TN) * LDT);
    const bf16_t* bufB = bufA + TM * LDT;
    v16bf afrag[2], bfrag[4];
#pragma unroll
    for (int fm = 0; fm < 2; fm++) {
      // A 16x32: lane-half h holds K = h*8..h*8+7 and 16+h*8..23+h*8
      const bf16_t* p = &bufA[(wm * 32 + fm * 16 + l16) * LDT + half * 8];
      afrag[fm] = cat8(*(const v8bf*)p, *(const v8bf*)(p + 16));
    }
#pragma unroll
    for (int fn = 0; fn < 4; fn++) {
      // B 32x16: lane-half h holds K = h*8..h*8+7 and 16+h*8..23+h*8
      const bf16_t* p = &bufB[(wn * 64 + fn * 16 + l16) * LDT + half * 8];
      bfrag[fn] = cat8(*(const v8bf*)p, *(const v8bf*)(p + 16));
    }
#pragma unroll
    for (int fm = 0; fm < 2; fm++)
#pragma unroll
      for (int fn = 0; fn < 4; fn++)
        acc[fm][fn] = __builtin_amdgcn_wmma_f32_16x16x32_bf16(
            false, afrag[fm], false, bfrag[fn], (short)0, acc[fm][fn], false,
            false);

#if !ASYNC_LDS
    if (more) {          // flush prefetched regs into buf[cur^1]
      *(v8bf*)(nb + sr0 * LDT + sc0) = a0;
      *(v8bf*)(nb + sr1 * LDT + sc1) = a1;
      *(v8bf*)(nb + (TM + sr0) * LDT + sc0) = b0;
      *(v8bf*)(nb + (TM + sr1) * LDT + sc1) = b1;
    }
#endif
    cur ^= 1;
  }

  // ---- epilogue: C/D layout -> lane l16 = col, VGPR g -> row (+8 for half) ----
#pragma unroll
  for (int fm = 0; fm < 2; fm++) {
#pragma unroll
    for (int fn = 0; fn < 4; fn++) {
      int col = nBase + wn * 64 + fn * 16 + l16;
#pragma unroll
      for (int g = 0; g < 8; g++) {
        int row = mBase + wm * 32 + fm * 16 + half * 8 + g;
        float v = acc[fm][fn][g];
        if constexpr (EPI == 0) {
          float r = (v + ep0[col]) * ep1[col];     // (u + b_in) * B_vec
          ((bf16_t*)outv)[(size_t)row * N + col] = (bf16_t)r;
        } else {
          float r = v + ep0[(size_t)row * N + col] * ep1[col];  // + x*D
          ((float*)outv)[(size_t)row * N + col] = r;
        }
      }
    }
  }
}

// ---------------------------------------------------------------------------
// Chunked linear recurrence  h_t = A*h_{t-1} + b_t  (per-channel scalar A).
// ---------------------------------------------------------------------------
__global__ void scan_carries(const bf16_t* __restrict__ bh,
                             const float* __restrict__ Ach,
                             float* __restrict__ carries, int d) {
  int gid = blockIdx.x * blockDim.x + threadIdx.x;  // = chunk*d + c
  int c = gid % d;
  int chunk = gid / d;
  float A = Ach[c];
  const bf16_t* p = bh + (size_t)chunk * CHUNK * d + c;
  float h = 0.f;
#pragma unroll 8
  for (int t = 0; t < CHUNK; ++t) h = fmaf(A, h, (float)p[(size_t)t * d]);
  carries[gid] = h;
}

__global__ void scan_combine(const float* __restrict__ Ach,
                             float* __restrict__ carries, int d,
                             int nchPerBatch) {
  int gid = blockIdx.x * blockDim.x + threadIdx.x;  // = batch*d + c
  int c = gid % d;
  int b = gid / d;
  float A = Ach[c];
  float AL = 1.f, p = A;
  int e = CHUNK;
  while (e) {                                   // AL = A^CHUNK
    if (e & 1) AL *= p;
    p *= p;
    e >>= 1;
  }
  float r = 0.f;
  for (int j = 0; j < nchPerBatch; ++j) {
    size_t idx = ((size_t)(b * nchPerBatch + j)) * d + c;
    float cj = carries[idx];
    carries[idx] = r;                           // overwrite with carry-in
    r = fmaf(AL, r, cj);
  }
}

__global__ void scan_fix(bf16_t* __restrict__ bh, const float* __restrict__ Ach,
                         const float* __restrict__ carries, int d) {
  int gid = blockIdx.x * blockDim.x + threadIdx.x;  // = chunk*d + c
  int c = gid % d;
  int chunk = gid / d;
  float A = Ach[c];
  float h = carries[gid];
  bf16_t* p = bh + (size_t)chunk * CHUNK * d + c;
#pragma unroll 8
  for (int t = 0; t < CHUNK; ++t) {
    h = fmaf(A, h, (float)p[(size_t)t * d]);
    p[(size_t)t * d] = (bf16_t)h;               // h overwrites b in place
  }
}

// ---------------------------------------------------------------------------
// Host launcher
// ---------------------------------------------------------------------------
extern "C" void kernel_launch(void* const* d_in, const int* in_sizes, int n_in,
                              void* d_out, int out_size, void* d_ws,
                              size_t ws_size, hipStream_t stream) {
  const float* x    = (const float*)d_in[0];
  const float* W_in = (const float*)d_in[1];
  const float* b_in = (const float*)d_in[2];
  const float* logA = (const float*)d_in[3];
  const float* Bvec = (const float*)d_in[4];
  const float* C    = (const float*)d_in[5];
  const float* D    = (const float*)d_in[6];
  float* out = (float*)d_out;

  const int d = in_sizes[2];          // 1024
  const int M = in_sizes[0] / d;      // B*T = 16384
  const int Bsz = 8;                  // reference setup
  const int T = M / Bsz;              // 2048
  const int nChunks = M / CHUNK;      // 128
  const int nchPerBatch = T / CHUNK;  // 16

  // Workspace carve-up (regions naturally 256B-aligned): ~68.5 MB total
  char* ws = (char*)d_ws;
  bf16_t* Wt = (bf16_t*)ws;  ws += (size_t)d * d * sizeof(bf16_t);        // 2 MB
  bf16_t* Ct = (bf16_t*)ws;  ws += (size_t)d * d * sizeof(bf16_t);        // 2 MB
  float* Ach = (float*)ws;   ws += (size_t)d * sizeof(float);             // 4 KB
  float* carries = (float*)ws; ws += (size_t)nChunks * d * sizeof(float); // .5 MB
  bf16_t* bh  = (bf16_t*)ws; ws += (size_t)M * d * sizeof(bf16_t);        // 32 MB
  bf16_t* xbf = (bf16_t*)ws;                                              // 32 MB

  // 1) prep
  long n4 = (long)M * d / 4;
  f32_to_bf16_vec<<<(int)((n4 + 255) / 256), 256, 0, stream>>>(
      (const float4*)x, (v4bf*)xbf, n4);
  dim3 tg(d / 64, d / 64);
  transpose_to_bf16<<<tg, 256, 0, stream>>>(W_in, Wt, d, d);
  transpose_to_bf16<<<tg, 256, 0, stream>>>(C, Ct, d, d);
  prep_A<<<(d + 255) / 256, 256, 0, stream>>>(logA, Ach, d);

  // 2) GEMM1: b = (x@W_in + b_in) * B_vec  -> bf16
  dim3 gg(d / TN, M / TM);
  ssm_gemm<0><<<gg, 256, 0, stream>>>(xbf, Wt, b_in, Bvec, bh, M, d, d);

  // 3) scan
  scan_carries<<<(nChunks * d) / 256, 256, 0, stream>>>(bh, Ach, carries, d);
  scan_combine<<<(Bsz * d) / 256, 256, 0, stream>>>(Ach, carries, d,
                                                    nchPerBatch);
  scan_fix<<<(nChunks * d) / 256, 256, 0, stream>>>(bh, Ach, carries, d);

  // 4) GEMM2: out = h@C + x*D  -> fp32
  ssm_gemm<1><<<gg, 256, 0, stream>>>(bh, Ct, x, D, out, M, d, d);
}
